// Decoder_20229295964336
// MI455X (gfx1250) — matrix operand order repaired and checked by reading, not compiled, not run on hardware
//
#include <hip/hip_runtime.h>

// ---------------- CDNA5 WMMA types ----------------
typedef __attribute__((ext_vector_type(16))) _Float16 v16h;
typedef __attribute__((ext_vector_type(8)))  _Float16 v8h;
typedef __attribute__((ext_vector_type(8)))  float    v8f;

// ---------------- model dims ----------------
// FC=128 META=128 P=256 H=1024 E=512 HA=128 WL=33 CP=8 K=17 T=512 B=64 F=256
#define DB      64
#define DT      512
#define DF      256
#define DFC     128
#define DMETA   128
#define DP      256
#define DH      1024
#define DE      512
#define DHA     128
#define DWL     33
#define DCP     8
#define DKC     17
#define DIN     640          // META + E
#define DPAD    16           // WL/2
#define DPADDING 24          // PAD + CP
#define DTPAD   544          // T + 2*PAD
#define DCUM    560          // T + 2*PADDING
#define DG4     4096         // 4*H
#define DKI1    896          // P + IN
#define DKI2    1664         // H + IN

// ---------------- device helpers ----------------
__device__ __forceinline__ v8f vzero8() {
  v8f r;
#pragma unroll
  for (int i = 0; i < 8; ++i) r[i] = 0.f;
  return r;
}

__device__ __forceinline__ float sigm(float x) { return 1.f / (1.f + __expf(-x)); }

// A fragment (16x32 f16, M x K). tile_base points at row 0 of the 16-row tile.
// Lane l<16 holds row l, K = k0..k0+15 (contiguous); lane l>=16 row l-16, K = k0+16..k0+31.
__device__ __forceinline__ v16h load_a_frag(const _Float16* __restrict__ tile_base,
                                            int lda, int k0, int lane) {
  int hsel = lane >> 4;
  int r = lane & 15;
  const _Float16* p = tile_base + (size_t)r * lda + k0 + 16 * hsel;
  return *(const v16h*)p;
}

// B fragment (32x16 f16, K x N) from transposed weights Wt stored N x K row-major.
// Lane l<16 holds col (col16+l), K = k0..k0+15 (contiguous 32B); lane>=16: K=k0+16..k0+31.
__device__ __forceinline__ v16h load_b_frag(const _Float16* __restrict__ Wt,
                                            int ldk, int col16, int k0, int lane) {
  int hsel = lane >> 4;
  int c = lane & 15;
  const _Float16* p = Wt + (size_t)(col16 + c) * ldk + k0 + 16 * hsel;
  return *(const v16h*)p;
}

__device__ __forceinline__ v8f wmma32(v16h a, v16h b, v8f c) {
  return __builtin_amdgcn_wmma_f32_16x16x32_f16(false, a, false, b, (short)0, c,
                                                false, false);
}

// ---------------- conversion kernels ----------------
__global__ void cvt_tr_kernel(const float* __restrict__ W, _Float16* __restrict__ Wt,
                              int Kd, int N) {
  int idx = blockIdx.x * 256 + threadIdx.x;
  if (idx >= Kd * N) return;
  int n = idx % N;
  int k = idx / N;
  Wt[(size_t)n * Kd + k] = (_Float16)W[idx];
}

__global__ void cvt_kernel(const float* __restrict__ s, _Float16* __restrict__ d, int n) {
  int idx = blockIdx.x * 256 + threadIdx.x;
  if (idx < n) d[idx] = (_Float16)s[idx];
}

// ---------------- init MLP ----------------
__global__ void init1_kernel(const float* __restrict__ meta,
                             const float* __restrict__ tokens,
                             const int* __restrict__ num_tokens,
                             const float* __restrict__ W1,   // 1152 x 640
                             const float* __restrict__ b1v,  // 640
                             float* __restrict__ hid) {      // 64 x 640
  int idx = blockIdx.x * 256 + threadIdx.x;
  if (idx >= DB * DIN) return;
  int col = idx % DIN;
  int b = idx / DIN;
  int nt = num_tokens[b];
  const float* tok0 = tokens + (size_t)b * DE;                       // tokens[0,b,:]
  const float* tokL = tokens + ((size_t)(nt - 1) * DB + b) * DE;     // tokens[nt-1,b,:]
  float acc = b1v[col];
  for (int k = 0; k < DMETA; ++k) acc += meta[b * DMETA + k] * W1[(size_t)k * DIN + col];
  for (int k = 0; k < DE; ++k)    acc += tok0[k] * W1[(size_t)(DMETA + k) * DIN + col];
  for (int k = 0; k < DE; ++k)    acc += tokL[k] * W1[(size_t)(DMETA + DE + k) * DIN + col];
  hid[idx] = acc > 0.f ? acc : 0.f;
}

__global__ void init2_kernel(const float* __restrict__ hid,
                             const float* __restrict__ W2,   // 640 x 1665
                             const float* __restrict__ b2v,
                             _Float16* __restrict__ xbuf,    // s=0 frame slot
                             float* __restrict__ init_cum,
                             _Float16* __restrict__ ctx_h,
                             _Float16* __restrict__ beg,
                             _Float16* __restrict__ endt) {
  const int NC = DFC + 1 + 3 * DE;  // 1665
  int idx = blockIdx.x * 256 + threadIdx.x;
  if (idx >= DB * NC) return;
  int col = idx % NC;
  int b = idx / NC;
  const float* h = hid + b * DIN;
  float acc = b2v[col];
  for (int k = 0; k < DIN; ++k) acc += h[k] * W2[(size_t)k * NC + col];
  if (col < DFC)            xbuf[(size_t)b * (DFC + DMETA) + col] = (_Float16)acc;
  else if (col == DFC)      init_cum[b] = acc;
  else if (col < DFC + 1 + DE)        ctx_h[b * DE + (col - DFC - 1)] = (_Float16)acc;
  else if (col < DFC + 1 + 2 * DE)    beg[b * DE + (col - DFC - 1 - DE)] = (_Float16)acc;
  else                                endt[b * DE + (col - DFC - 1 - 2 * DE)] = (_Float16)acc;
}

__global__ void build_x_kernel(const float* __restrict__ target_frames,  // 256,64,128
                               const float* __restrict__ meta,
                               _Float16* __restrict__ xbuf) {            // 256,64,256
  int idx = blockIdx.x * 256 + threadIdx.x;
  if (idx >= DF * DB * (DFC + DMETA)) return;
  int j = idx & 255;
  int r = idx >> 8;
  int b = r & 63;
  int s = r >> 6;
  if (j >= DFC)      xbuf[idx] = (_Float16)meta[b * DMETA + (j - DFC)];
  else if (s > 0)    xbuf[idx] = (_Float16)target_frames[((size_t)(s - 1) * DB + b) * DFC + j];
  // s==0, j<DFC written by init2_kernel
}

__global__ void misc_init_kernel(const float* __restrict__ init_cum,
                                 float* __restrict__ cum, float* __restrict__ c1,
                                 float* __restrict__ c2,
                                 _Float16* __restrict__ h1a, _Float16* __restrict__ h1b,
                                 _Float16* __restrict__ h2a, _Float16* __restrict__ h2b,
                                 int* __restrict__ wsidx) {
  int idx = blockIdx.x * 256 + threadIdx.x;
  if (idx < DB * DCUM) {
    int t = idx % DCUM;
    int b = idx / DCUM;
    cum[idx] = (t < DPADDING) ? fabsf(init_cum[b]) : 0.f;
  }
  if (idx < DB * DH) {
    c1[idx] = 0.f; c2[idx] = 0.f;
    h1a[idx] = (_Float16)0.f; h1b[idx] = (_Float16)0.f;
    h2a[idx] = (_Float16)0.f; h2b[idx] = (_Float16)0.f;
  }
  if (idx < DB) wsidx[idx] = 0;
}

// ---------------- padded token tensor (B, TPAD, E) f16 ----------------
__global__ void build_ptoks_kernel(const float* __restrict__ tokens,  // T,B,E
                                   const int* __restrict__ num_tokens,
                                   const _Float16* __restrict__ beg,
                                   const _Float16* __restrict__ endt,
                                   _Float16* __restrict__ ptoks) {
  size_t idx = (size_t)blockIdx.x * 256 + threadIdx.x;
  if (idx >= (size_t)DB * DTPAD * DE) return;
  int e = (int)(idx & (DE - 1));
  size_t r = idx >> 9;
  int t = (int)(r % DTPAD);
  int b = (int)(r / DTPAD);
  int nt = num_tokens[b];
  float val = 0.f;
  int tt = t - DPAD;
  if (tt >= 0 && tt < DT) val = tokens[((size_t)tt * DB + b) * DE + e];
  bool oldm = (t >= DPAD) && (t < DPAD + nt);
  bool newm = t < nt + 2 * DPAD;
  _Float16 hv;
  if (t < DPAD)           hv = beg[b * DE + e];
  else if (oldm != newm)  hv = endt[b * DE + e];
  else                    hv = (_Float16)val;
  ptoks[idx] = hv;
}

// ---------------- generic WMMA GEMM: C(MxN) = A(MxK) * B(KxN) ----------------
// A row-major f16 (lda==Kd), Bt = B transposed (N x K) f16. Optional bias+relu, f16 or f32 out.
__global__ __launch_bounds__(256) void gemm_kernel(
    const _Float16* __restrict__ A, const _Float16* __restrict__ Bt, int M, int N,
    int Kd, const float* __restrict__ bias, int relu, float* __restrict__ Cf,
    _Float16* __restrict__ Ch) {
  int lane = threadIdx.x & 31;
  int wid = blockIdx.x * 8 + (threadIdx.x >> 5);
  int ntn = N >> 4;
  int total = (M >> 4) * ntn;
  if (wid >= total) return;
  int mt = wid / ntn;
  int nt = wid % ntn;
  v8f acc = vzero8();
  const _Float16* Atile = A + (size_t)mt * 16 * Kd;
  for (int k0 = 0; k0 < Kd; k0 += 32) {
    __builtin_prefetch(Bt + (size_t)(nt * 16 + (lane & 15)) * Kd + k0 + 128, 0, 1);
    v16h a = load_a_frag(Atile, Kd, k0, lane);
    v16h b = load_b_frag(Bt, Kd, nt * 16, k0, lane);
    acc = wmma32(a, b, acc);
  }
  int hsel = lane >> 4;
  int c = lane & 15;
  int coln = nt * 16 + c;
  float bv = bias ? bias[coln] : 0.f;
#pragma unroll
  for (int j = 0; j < 8; ++j) {
    int row = mt * 16 + j + 8 * hsel;
    float vce = acc[j] + bv;
    if (relu && vce < 0.f) vce = 0.f;
    if (Ch) Ch[(size_t)row * N + coln] = (_Float16)vce;
    else    Cf[(size_t)row * N + coln] = vce;
  }
}

// 3-segment A GEMM (for final frame projection): A = [A0 | A1 | A2], A2 rows broadcast mod 64.
__global__ __launch_bounds__(256) void gemm3_kernel(
    const _Float16* __restrict__ A0, int L0, const _Float16* __restrict__ A1, int L1,
    const _Float16* __restrict__ A2, int L2, const _Float16* __restrict__ Bt, int M,
    int N, const float* __restrict__ bias, float* __restrict__ Cf) {
  int lane = threadIdx.x & 31;
  int wid = blockIdx.x * 8 + (threadIdx.x >> 5);
  int ntn = N >> 4;
  int total = (M >> 4) * ntn;
  if (wid >= total) return;
  int mt = wid / ntn;
  int nt = wid % ntn;
  int Kd = L0 + L1 + L2;
  v8f acc = vzero8();
  const _Float16* A0t = A0 + (size_t)mt * 16 * L0;
  const _Float16* A1t = A1 + (size_t)mt * 16 * L1;
  const _Float16* A2t = A2 + (size_t)(mt & 3) * 16 * L2;  // row & 63
  for (int k0 = 0; k0 < Kd; k0 += 32) {
    const _Float16* ap; int lda, kk;
    if (k0 < L0)            { ap = A0t; lda = L0; kk = k0; }
    else if (k0 < L0 + L1)  { ap = A1t; lda = L1; kk = k0 - L0; }
    else                    { ap = A2t; lda = L2; kk = k0 - L0 - L1; }
    v16h a = load_a_frag(ap, lda, kk, lane);
    v16h b = load_b_frag(Bt, Kd, nt * 16, k0, lane);
    acc = wmma32(a, b, acc);
  }
  int hsel = lane >> 4;
  int c = lane & 15;
  int coln = nt * 16 + c;
  float bv = bias ? bias[coln] : 0.f;
#pragma unroll
  for (int j = 0; j < 8; ++j) {
    int row = mt * 16 + j + 8 * hsel;
    Cf[(size_t)row * N + coln] = acc[j] + bv;
  }
}

// ---------------- LSTM gate kernel (shared by LSTM1 & LSTM2) ----------------
// One block handles 32 hidden columns; 8 waves = 4 gates x 2 column halves.
// A = [A0 | A1 | A2] (64 x Ki), then h_prev (64 x H) against Wh. Cell update in LDS.
__global__ __launch_bounds__(256) void lstm_gates_kernel(
    const _Float16* __restrict__ A0, int L0, const _Float16* __restrict__ A1, int L1,
    const _Float16* __restrict__ A2, int L2, const _Float16* __restrict__ Wi_t,  // 4H x Ki
    const _Float16* __restrict__ Hprev,                                          // 64 x H
    const _Float16* __restrict__ Wh_t,                                           // 4H x H
    const float* __restrict__ bias,                                              // 4H
    float* __restrict__ c_state,                                                 // 64 x H
    _Float16* __restrict__ h_next,                                               // 64 x H
    _Float16* __restrict__ h_seq) {                                              // 64 x H
  __shared__ float gbuf[4][64][32];
  int lane = threadIdx.x & 31;
  int wave = threadIdx.x >> 5;
  int gate = wave >> 1;
  int nhalf = wave & 1;
  int hcol0 = blockIdx.x * 32;
  int col = gate * DH + hcol0 + nhalf * 16;
  int Ki = L0 + L1 + L2;

  v8f acc[4];
#pragma unroll
  for (int m = 0; m < 4; ++m) acc[m] = vzero8();

  // input contribution
  for (int k0 = 0; k0 < Ki; k0 += 32) {
    const _Float16* ap; int lda, kk;
    if (k0 < L0)            { ap = A0; lda = L0; kk = k0; }
    else if (k0 < L0 + L1)  { ap = A1; lda = L1; kk = k0 - L0; }
    else                    { ap = A2; lda = L2; kk = k0 - L0 - L1; }
    __builtin_prefetch(Wi_t + (size_t)(col + (lane & 15)) * Ki + k0 + 128, 0, 1);
    v16h b = load_b_frag(Wi_t, Ki, col, k0, lane);
#pragma unroll
    for (int m = 0; m < 4; ++m) {
      v16h a = load_a_frag(ap + (size_t)m * 16 * lda, lda, kk, lane);
      acc[m] = wmma32(a, b, acc[m]);
    }
  }
  // recurrent contribution
  for (int k0 = 0; k0 < DH; k0 += 32) {
    __builtin_prefetch(Wh_t + (size_t)(col + (lane & 15)) * DH + k0 + 128, 0, 1);
    v16h b = load_b_frag(Wh_t, DH, col, k0, lane);
#pragma unroll
    for (int m = 0; m < 4; ++m) {
      v16h a = load_a_frag(Hprev + (size_t)m * 16 * DH, DH, k0, lane);
      acc[m] = wmma32(a, b, acc[m]);
    }
  }

  int hsel = lane >> 4;
  int c = lane & 15;
  float bv = bias[col + c];
#pragma unroll
  for (int m = 0; m < 4; ++m)
#pragma unroll
    for (int j = 0; j < 8; ++j)
      gbuf[gate][m * 16 + j + 8 * hsel][nhalf * 16 + c] = acc[m][j] + bv;
  __syncthreads();

  for (int idx = threadIdx.x; idx < 64 * 32; idx += 256) {
    int rowb = idx >> 5;
    int cl = idx & 31;
    int hidx = hcol0 + cl;
    float ig = gbuf[0][rowb][cl];
    float fg = gbuf[1][rowb][cl];
    float gg = gbuf[2][rowb][cl];
    float og = gbuf[3][rowb][cl];
    size_t off = (size_t)rowb * DH + hidx;
    float cn = sigm(fg) * c_state[off] + sigm(ig) * tanhf(gg);
    float hn = sigm(og) * tanhf(cn);
    c_state[off] = cn;
    _Float16 hh = (_Float16)hn;
    h_next[off] = hh;
    h_seq[off] = hh;
  }
}

// ---------------- attention step: one block per batch row ----------------
__global__ __launch_bounds__(256) void attn_kernel(
    const float* __restrict__ q,        // 64 x 128
    const float* __restrict__ keys,     // 64 x 544 x 128
    const _Float16* __restrict__ ptoks, // 64 x 544 x 512
    const float* __restrict__ Wloc,     // 17 x 128
    const float* __restrict__ vvec,     // 128
    const int* __restrict__ num_tokens,
    float* __restrict__ cum,            // 64 x 560
    int* __restrict__ wsidx,            // 64
    _Float16* __restrict__ ctx_h,       // 64 x 512 (recurrent)
    _Float16* __restrict__ ctxs_seq,    // slot 64 x 512
    float* __restrict__ out_align,      // slot 64 x 512
    float* __restrict__ out_wss) {      // slot 64
  __shared__ float sh_loc[DWL + 2 * DCP];  // 49
  __shared__ float sh_score[DWL];
  __shared__ float sh_align[DWL];
  __shared__ float sh_q[DHA];
  __shared__ int sh_arg;
  int b = blockIdx.x;
  int tid = threadIdx.x;
  int wsb = wsidx[b];
  int nt = num_tokens[b];
  if (tid < DHA) sh_q[tid] = q[b * DHA + tid];
  if (tid < DWL + 2 * DCP) sh_loc[tid] = cum[b * DCUM + wsb + tid];
  if (tid < DWL) sh_score[tid] = 0.f;
  __syncthreads();

  for (int idx = tid; idx < DWL * DHA; idx += 256) {
    int w = idx >> 7;
    int h = idx & (DHA - 1);
    float lf = 0.f;
#pragma unroll
    for (int k = 0; k < DKC; ++k) lf += sh_loc[w + k] * Wloc[k * DHA + h];
    float kv = keys[((size_t)b * DTPAD + wsb + w) * DHA + h];
    float t = tanhf(kv + sh_q[h] + lf);
    atomicAdd(&sh_score[w], vvec[h] * t);
  }
  __syncthreads();

  if (tid == 0) {
    float mx = -1e30f;
    int am = 0;
    for (int w = 0; w < DWL; ++w) {
      int tpos = wsb + w;
      float sc = (tpos < nt + 2 * DPAD) ? sh_score[w] : -1e9f;
      sh_score[w] = sc;
      if (sc > mx) { mx = sc; am = w; }
    }
    float sum = 0.f;
    for (int w = 0; w < DWL; ++w) {
      float e = __expf(sh_score[w] - mx);
      sh_align[w] = e;
      sum += e;
    }
    float inv = 1.f / sum;
    for (int w = 0; w < DWL; ++w) sh_align[w] *= inv;
    sh_arg = am;
  }
  __syncthreads();

  // context vector
  for (int e = tid; e < DE; e += 256) {
    float a = 0.f;
    for (int w = 0; w < DWL; ++w)
      a += sh_align[w] * (float)ptoks[((size_t)b * DTPAD + wsb + w) * DE + e];
    _Float16 hv = (_Float16)a;
    ctx_h[b * DE + e] = hv;
    ctxs_seq[b * DE + e] = hv;
  }
  // cumulative attention update (widx + CP)
  if (tid < DWL) cum[b * DCUM + wsb + tid + DCP] += sh_align[tid];

  // masked alignment output slice
  for (int t = tid; t < DT; t += 256) out_align[b * DT + t] = 0.f;
  __syncthreads();
  if (tid < DWL) {
    int t = wsb + tid - DPAD;
    if (t >= 0 && t < DT && t < nt) out_align[b * DT + t] = sh_align[tid];
  }
  if (tid == 0) {
    int nws = wsb + sh_arg - DPAD;
    if (nws < 0) nws = 0;
    if (nws > DTPAD - DWL) nws = DTPAD - DWL;
    wsidx[b] = nws;
    out_wss[b] = (float)nws;
  }
}

// ---------------- stop token GEMV ----------------
__global__ __launch_bounds__(256) void stop_kernel(
    const _Float16* __restrict__ hs, const _Float16* __restrict__ ctxs,
    const _Float16* __restrict__ meta, const float* __restrict__ Wsv,
    const float* __restrict__ bsv, float* __restrict__ out) {
  int lane = threadIdx.x & 31;
  int row = blockIdx.x * 8 + (threadIdx.x >> 5);
  if (row >= DF * DB) return;
  const _Float16* h = hs + (size_t)row * DH;
  const _Float16* cx = ctxs + (size_t)row * DE;
  const _Float16* mt = meta + (size_t)(row & 63) * DMETA;
  float acc = 0.f;
  for (int k = lane; k < DH; k += 32)    acc += (float)h[k] * Wsv[k];
  for (int k = lane; k < DE; k += 32)    acc += (float)cx[k] * Wsv[DH + k];
  for (int k = lane; k < DMETA; k += 32) acc += (float)mt[k] * Wsv[DH + DE + k];
  for (int off = 16; off > 0; off >>= 1) acc += __shfl_down(acc, off, 32);
  if (lane == 0) out[row] = acc + bsv[0];
}

// ---------------- host launcher ----------------
static inline int cdiv(int a, int b) { return (a + b - 1) / b; }

extern "C" void kernel_launch(void* const* d_in, const int* in_sizes, int n_in,
                              void* d_out, int out_size, void* d_ws, size_t ws_size,
                              hipStream_t stream) {
  (void)in_sizes; (void)n_in; (void)out_size; (void)ws_size;
  const float* tokens        = (const float*)d_in[0];
  const float* seq_metadata  = (const float*)d_in[1];
  const float* target_frames = (const float*)d_in[2];
  const int*   num_tokens    = (const int*)d_in[3];
  const float* W_init1 = (const float*)d_in[4];
  const float* b_init1 = (const float*)d_in[5];
  const float* W_init2 = (const float*)d_in[6];
  const float* b_init2 = (const float*)d_in[7];
  const float* Wp1 = (const float*)d_in[8];
  const float* bp1 = (const float*)d_in[9];
  const float* Wp2 = (const float*)d_in[10];
  const float* bp2 = (const float*)d_in[11];
  const float* Wi1 = (const float*)d_in[12];
  const float* Wh1 = (const float*)d_in[13];
  const float* b1  = (const float*)d_in[14];
  const float* Wi2 = (const float*)d_in[15];
  const float* Wh2 = (const float*)d_in[16];
  const float* b2  = (const float*)d_in[17];
  const float* We  = (const float*)d_in[18];
  const float* Wq  = (const float*)d_in[19];
  const float* Wloc = (const float*)d_in[20];
  const float* vvec = (const float*)d_in[21];
  const float* Wo  = (const float*)d_in[22];
  const float* bo  = (const float*)d_in[23];
  const float* Wsp = (const float*)d_in[24];
  const float* bsp = (const float*)d_in[25];
  float* out = (float*)d_out;

  // output offsets (frames, stop, align_out, wss)
  const size_t o_stop  = (size_t)DF * DB * DFC;           // 2,097,152
  const size_t o_align = o_stop + (size_t)DF * DB;        // +16,384
  const size_t o_wss   = o_align + (size_t)DF * DB * DT;  // +8,388,608

  // ---- deterministic workspace carve ----
  char* base = (char*)d_ws;
  size_t off = 0;
  auto take = [&](size_t bytes) -> char* {
    char* p = base + off;
    off += (bytes + 255) & ~(size_t)255;
    return p;
  };
  _Float16* Wi1_t = (_Float16*)take((size_t)DG4 * DKI1 * 2);
  _Float16* Wh1_t = (_Float16*)take((size_t)DG4 * DH * 2);
  _Float16* Wi2_t = (_Float16*)take((size_t)DG4 * DKI2 * 2);
  _Float16* Wh2_t = (_Float16*)take((size_t)DG4 * DH * 2);
  _Float16* Wp1_t = (_Float16*)take((size_t)DP * (DFC + DMETA) * 2);
  _Float16* Wp2_t = (_Float16*)take((size_t)DP * DP * 2);
  _Float16* We_t  = (_Float16*)take((size_t)DHA * DE * 2);
  _Float16* Wq_t  = (_Float16*)take((size_t)DHA * DH * 2);
  _Float16* Wo_t  = (_Float16*)take((size_t)DFC * DKI2 * 2);
  _Float16* meta_h = (_Float16*)take((size_t)DB * DMETA * 2);
  _Float16* xbuf  = (_Float16*)take((size_t)DF * DB * (DFC + DMETA) * 2);
  _Float16* pre1  = (_Float16*)take((size_t)DF * DB * DP * 2);
  _Float16* pre   = (_Float16*)take((size_t)DF * DB * DP * 2);
  _Float16* beg   = (_Float16*)take((size_t)DB * DE * 2);
  _Float16* endt  = (_Float16*)take((size_t)DB * DE * 2);
  _Float16* ptoks = (_Float16*)take((size_t)DB * DTPAD * DE * 2);
  _Float16* ctx_h = (_Float16*)take((size_t)DB * DE * 2);
  _Float16* ctxs  = (_Float16*)take((size_t)DF * DB * DE * 2);
  _Float16* h1a   = (_Float16*)take((size_t)DB * DH * 2);
  _Float16* h1b   = (_Float16*)take((size_t)DB * DH * 2);
  _Float16* h2a   = (_Float16*)take((size_t)DB * DH * 2);
  _Float16* h2b   = (_Float16*)take((size_t)DB * DH * 2);
  _Float16* hs    = (_Float16*)take((size_t)DF * DB * DH * 2);
  _Float16* l2out = (_Float16*)take((size_t)DF * DB * DH * 2);
  float* keys     = (float*)take((size_t)DB * DTPAD * DHA * 4);
  float* cum      = (float*)take((size_t)DB * DCUM * 4);
  float* qbuf     = (float*)take((size_t)DB * DHA * 4);
  float* c1       = (float*)take((size_t)DB * DH * 4);
  float* c2       = (float*)take((size_t)DB * DH * 4);
  float* hid      = (float*)take((size_t)DB * DIN * 4);
  float* init_cum = (float*)take((size_t)DB * 4);
  int*   wsidx    = (int*)take((size_t)DB * 4);

  // ---- weight conversion (f32 -> f16 transposed N x K) ----
  cvt_tr_kernel<<<cdiv(DKI1 * DG4, 256), 256, 0, stream>>>(Wi1, Wi1_t, DKI1, DG4);
  cvt_tr_kernel<<<cdiv(DH * DG4, 256), 256, 0, stream>>>(Wh1, Wh1_t, DH, DG4);
  cvt_tr_kernel<<<cdiv(DKI2 * DG4, 256), 256, 0, stream>>>(Wi2, Wi2_t, DKI2, DG4);
  cvt_tr_kernel<<<cdiv(DH * DG4, 256), 256, 0, stream>>>(Wh2, Wh2_t, DH, DG4);
  cvt_tr_kernel<<<cdiv((DFC + DMETA) * DP, 256), 256, 0, stream>>>(Wp1, Wp1_t, DFC + DMETA, DP);
  cvt_tr_kernel<<<cdiv(DP * DP, 256), 256, 0, stream>>>(Wp2, Wp2_t, DP, DP);
  cvt_tr_kernel<<<cdiv(DE * DHA, 256), 256, 0, stream>>>(We, We_t, DE, DHA);
  cvt_tr_kernel<<<cdiv(DH * DHA, 256), 256, 0, stream>>>(Wq, Wq_t, DH, DHA);
  cvt_tr_kernel<<<cdiv(DKI2 * DFC, 256), 256, 0, stream>>>(Wo, Wo_t, DKI2, DFC);
  cvt_kernel<<<cdiv(DB * DMETA, 256), 256, 0, stream>>>(seq_metadata, meta_h, DB * DMETA);

  // ---- init path ----
  init1_kernel<<<cdiv(DB * DIN, 256), 256, 0, stream>>>(seq_metadata, tokens, num_tokens,
                                                        W_init1, b_init1, hid);
  init2_kernel<<<cdiv(DB * (DFC + 1 + 3 * DE), 256), 256, 0, stream>>>(
      hid, W_init2, b_init2, xbuf, init_cum, ctx_h, beg, endt);
  build_x_kernel<<<cdiv(DF * DB * (DFC + DMETA), 256), 256, 0, stream>>>(
      target_frames, seq_metadata, xbuf);
  misc_init_kernel<<<cdiv(DB * DCUM, 256), 256, 0, stream>>>(init_cum, cum, c1, c2, h1a,
                                                             h1b, h2a, h2b, wsidx);
  build_ptoks_kernel<<<cdiv(DB * DTPAD * DE, 256), 256, 0, stream>>>(tokens, num_tokens,
                                                                     beg, endt, ptoks);

  // ---- keys = ptoks @ We : (B*TPAD x E) @ (E x HA) ----
  {
    int tiles = (DB * DTPAD / 16) * (DHA / 16);
    gemm_kernel<<<cdiv(tiles, 8), 256, 0, stream>>>(ptoks, We_t, DB * DTPAD, DHA, DE,
                                                    nullptr, 0, keys, nullptr);
  }
  // ---- prenet: relu(relu(x@Wp1+bp1)@Wp2+bp2) ----
  {
    int tiles = (DF * DB / 16) * (DP / 16);
    gemm_kernel<<<cdiv(tiles, 8), 256, 0, stream>>>(xbuf, Wp1_t, DF * DB, DP,
                                                    DFC + DMETA, bp1, 1, nullptr, pre1);
    gemm_kernel<<<cdiv(tiles, 8), 256, 0, stream>>>(pre1, Wp2_t, DF * DB, DP, DP, bp2, 1,
                                                    nullptr, pre);
  }

  // ---- main attention/LSTM1 scan ----
  for (int s = 0; s < DF; ++s) {
    _Float16* hprev = (s & 1) ? h1b : h1a;
    _Float16* hnext = (s & 1) ? h1a : h1b;
    lstm_gates_kernel<<<DH / 32, 256, 0, stream>>>(
        pre + (size_t)s * DB * DP, DP, ctx_h, DE, meta_h, DMETA, Wi1_t, hprev, Wh1_t, b1,
        c1, hnext, hs + (size_t)s * DB * DH);
    int qtiles = (DB / 16) * (DHA / 16);
    gemm_kernel<<<cdiv(qtiles, 8), 256, 0, stream>>>(hnext, Wq_t, DB, DHA, DH, nullptr, 0,
                                                     qbuf, nullptr);
    attn_kernel<<<DB, 256, 0, stream>>>(
        qbuf, keys, ptoks, Wloc, vvec, num_tokens, cum, wsidx, ctx_h,
        ctxs + (size_t)s * DB * DE, out + o_align + (size_t)s * DB * DT,
        out + o_wss + (size_t)s * DB);
  }

  // ---- stop token ----
  stop_kernel<<<cdiv(DF * DB, 8), 256, 0, stream>>>(hs, ctxs, meta_h, Wsp, bsp,
                                                    out + o_stop);

  // ---- LSTM2 scan ----
  for (int s = 0; s < DF; ++s) {
    _Float16* hprev = (s & 1) ? h2b : h2a;
    _Float16* hnext = (s & 1) ? h2a : h2b;
    lstm_gates_kernel<<<DH / 32, 256, 0, stream>>>(
        hs + (size_t)s * DB * DH, DH, ctxs + (size_t)s * DB * DE, DE, meta_h, DMETA,
        Wi2_t, hprev, Wh2_t, b2, c2, hnext, l2out + (size_t)s * DB * DH);
  }

  // ---- frames = [l2out|ctxs|meta] @ Wo + bo ----
  {
    int tiles = (DF * DB / 16) * (DFC / 16);
    gemm3_kernel<<<cdiv(tiles, 8), 256, 0, stream>>>(l2out, DH, ctxs, DE, meta_h, DMETA,
                                                     Wo_t, DF * DB, DFC, bo, out);
  }
}
